// NormalizeCLAHE2_31258771980571
// MI455X (gfx1250) — matrix operand order repaired and checked by reading, not compiled, not run on hardware
//
#include <hip/hip_runtime.h>
#include <stdint.h>

// ---------------- problem constants (match reference setup_inputs) ----------
#define BATCH   8
#define H_IMG   2048
#define W_IMG   2048
#define NPIX    (BATCH * H_IMG * W_IMG)     // 33,554,432
#define NV4     (NPIX / 4)                  // 8,388,608 float4
#define TILE    256                         // 2048 / 8
#define GH      8
#define GW      8
#define NTILES  (BATCH * GH * GW)           // 512
#define NBINS   256
#define CLIPMAX 256u                        // clip_limit*pixels//bins = 65536/256
#define CHUNK   16                          // rows per TDM chunk (16*256*4B = 16KB)
#define NCHUNK  (TILE / CHUNK)              // 16

// ---------------- workspace layout (float offsets) --------------------------
#define WS_P1MIN 0
#define WS_P1MAX 1024
#define WS_MN1   2048
#define WS_MX1   2049
#define WS_LUT   2056
#define WS_P2MIN (WS_LUT + NTILES * NBINS)  // 133128
#define WS_P2MAX (WS_P2MIN + 2048)
#define WS_MN2   (WS_P2MIN + 4096)
#define WS_MX2   (WS_MN2 + 1)
// total ~137,226 floats (~549 KB) of d_ws used

typedef __attribute__((ext_vector_type(4)))  unsigned int v4u;
typedef __attribute__((ext_vector_type(8)))  int          v8i;
typedef __attribute__((ext_vector_type(4)))  int          v4i;
typedef __attribute__((ext_vector_type(16))) _Float16     v16h;
typedef __attribute__((ext_vector_type(8)))  float        v8f;

// ---------------- helpers ----------------------------------------------------
__device__ inline int clampi(int v, int lo, int hi) {
  return v < lo ? lo : (v > hi ? hi : v);
}

__device__ inline void block_reduce_minmax(float lmn, float lmx,
                                           float* pmn, float* pmx) {
  __shared__ float smn[256];
  __shared__ float smx[256];
  const int t = threadIdx.x;
  smn[t] = lmn; smx[t] = lmx;
  __syncthreads();
  for (int s = 128; s > 0; s >>= 1) {
    if (t < s) {
      smn[t] = fminf(smn[t], smn[t + s]);
      smx[t] = fmaxf(smx[t], smx[t + s]);
    }
    __syncthreads();
  }
  if (t == 0) { *pmn = smn[0]; *pmx = smx[0]; }
}

// map normalized input -> CLAHE bin index (matches reference exactly)
__device__ inline int pix_bin(float x, float mn, float inv) {
  float t = (x - mn) * inv;              // normalize_in_range
  float v = 2.5f * __log2f(1.0f + t);    // adjust_log, LOG_GAIN=2.5
  v = fminf(fmaxf(v, 0.0f), 1.0f);       // clip [0,1]
  int b = (int)(v * 256.0f);
  return b > 255 ? 255 : b;
}

// low 32 bits of a generic pointer == LDS byte offset (flat aperture rule)
__device__ inline unsigned lds_byte_off(const void* p) {
  return (unsigned)(unsigned long long)(uintptr_t)p;
}

// ---------------- K1: per-block min/max of input -----------------------------
__global__ __launch_bounds__(256) void k_minmax_in(const float* __restrict__ in,
                                                   float* __restrict__ ws) {
  float lmn = 3.402823466e38f, lmx = -3.402823466e38f;
  const float4* in4 = (const float4*)in;
  for (unsigned i = blockIdx.x * 256u + threadIdx.x; i < NV4; i += gridDim.x * 256u) {
    float4 v = in4[i];
    lmn = fminf(lmn, fminf(fminf(v.x, v.y), fminf(v.z, v.w)));
    lmx = fmaxf(lmx, fmaxf(fmaxf(v.x, v.y), fmaxf(v.z, v.w)));
  }
  block_reduce_minmax(lmn, lmx, &ws[WS_P1MIN + blockIdx.x], &ws[WS_P1MAX + blockIdx.x]);
}

// ---------------- K2: finalize input min/max (1024 partials) -----------------
__global__ __launch_bounds__(256) void k_final1(float* __restrict__ ws) {
  const int t = threadIdx.x;
  float lmn = 3.402823466e38f, lmx = -3.402823466e38f;
  for (int i = t; i < 1024; i += 256) {
    lmn = fminf(lmn, ws[WS_P1MIN + i]);
    lmx = fmaxf(lmx, ws[WS_P1MAX + i]);
  }
  block_reduce_minmax(lmn, lmx, &ws[WS_MN1], &ws[WS_MX1]);
}

// ---------------- K3: per-tile histogram -> clip -> WMMA CDF -> LUT ----------
// One workgroup per tile (512 WGs, 256 threads = 8 waves).
// Tile pixels staged into LDS in 16-row chunks by the Tensor Data Mover
// (double-buffered, s_wait_tensorcnt), histogram via ds_add_u32 atomics,
// 256-bin cumsum via one v_wmma_f32_16x16x32_f16 against an upper-triangular
// ones matrix (bin counts <= 513 are exact in f16; accumulation exact in f32).
__global__ __launch_bounds__(256) void k_tile_lut(const float* __restrict__ in,
                                                  float* __restrict__ ws) {
  __shared__ float    tbuf[2][CHUNK * TILE];   // 2 x 16KB staging
  __shared__ unsigned hist[NBINS];
  __shared__ float    cum[NBINS];              // histf, then row-prefixes
  __shared__ float    off16[16];
  __shared__ unsigned red[NBINS];

  const int t   = threadIdx.x;
  const int blk = blockIdx.x;                  // 0..511
  const int b   = blk >> 6;
  const int ty  = (blk >> 3) & 7;
  const int tx  = blk & 7;

  const float mn  = ws[WS_MN1];
  const float mx  = ws[WS_MX1];
  const float inv = 1.0f / (mx - mn);

  hist[t] = 0u;
  __syncthreads();

  const size_t tileElemBase = (size_t)b * (size_t)(H_IMG * W_IMG)
                            + (size_t)ty * TILE * W_IMG
                            + (size_t)tx * TILE;

#if defined(__gfx1250__)
  const unsigned long long gbase =
      (unsigned long long)(uintptr_t)in + (unsigned long long)tileElemBase * 4ull;
  const unsigned long long chunkStride = (unsigned long long)CHUNK * W_IMG * 4ull;

  // D# groups per CDNA5 ISA (2D tensor, data_size=4B)
  auto tdm_issue = [&](unsigned long long gaddr, unsigned ldsByte) {
    v4u g0 = { 1u,                                        // count=1, no gather
               ldsByte,                                   // lds_addr
               (unsigned)(gaddr & 0xFFFFFFFFull),         // global_addr lo
               (unsigned)((gaddr >> 32) & 0x01FFFFFFull)  // global_addr hi
                 | (2u << 30) };                          // type=2 (image)
    v8i g1 = { (int)(2u << 16),                 // data_size=2 (4 bytes)
               (int)(2048u << 16),              // tensor_dim0 = 2048 (lo16 @48)
               (int)(2048u << 16),              // dim0 hi=0 | tensor_dim1 lo16
               (int)(256u << 16),               // dim1 hi=0 | tile_dim0 = 256
               (int)CHUNK,                      // tile_dim1 = 16, tile_dim2 = 0
               2048,                            // tensor_dim0_stride lo32
               0,                               // stride0 hi | stride1 lo16
               64 };                            // tensor_dim1_stride hi (2048*2048)
    v4i gz = { 0, 0, 0, 0 };
#if defined(__clang_major__) && (__clang_major__ >= 23)
    v8i g4 = { 0, 0, 0, 0, 0, 0, 0, 0 };
    __builtin_amdgcn_tensor_load_to_lds(g0, g1, gz, gz, g4, 0);
#else
    __builtin_amdgcn_tensor_load_to_lds(g0, g1, gz, gz, 0);
#endif
  };

  if (t < 32) tdm_issue(gbase, lds_byte_off(&tbuf[0][0]));  // prime chunk 0
  for (int c = 0; c < NCHUNK; ++c) {
    if (t < 32) {                                   // wave 0 drives the TDM
      if (c + 1 < NCHUNK) {
        tdm_issue(gbase + (unsigned long long)(c + 1) * chunkStride,
                  lds_byte_off(&tbuf[(c + 1) & 1][0]));
        __builtin_amdgcn_s_wait_tensorcnt(1);       // chunk c retired (in-order)
      } else {
        __builtin_amdgcn_s_wait_tensorcnt(0);
      }
    }
    __syncthreads();
    const float* buf = tbuf[c & 1];
    for (int j = 0; j < CHUNK; ++j) {
      int bin = pix_bin(buf[j * TILE + t], mn, inv);
      atomicAdd(&hist[bin], 1u);                    // ds_add_u32
    }
    __syncthreads();                                // before buffer reuse
  }
#else
  // host-pass / non-gfx1250 fallback: cooperative copy, same structure
  for (int c = 0; c < NCHUNK; ++c) {
    for (int j = 0; j < CHUNK; ++j)
      tbuf[0][j * TILE + t] = in[tileElemBase + (size_t)(c * CHUNK + j) * W_IMG + t];
    __syncthreads();
    for (int j = 0; j < CHUNK; ++j) {
      int bin = pix_bin(tbuf[0][j * TILE + t], mn, inv);
      atomicAdd(&hist[bin], 1u);
    }
    __syncthreads();
  }
#endif

  // ---- clip + uniform redistribution (integer, exact) ----
  red[t] = hist[t] > CLIPMAX ? hist[t] - CLIPMAX : 0u;
  __syncthreads();
  for (int s = 128; s > 0; s >>= 1) {
    if (t < s) red[t] += red[t + s];
    __syncthreads();
  }
  const unsigned excess = red[0];
  const unsigned redist = excess >> 8;
  const unsigned resid  = excess - (redist << 8);
  unsigned hv = (hist[t] > CLIPMAX ? CLIPMAX : hist[t]) + redist
              + ((unsigned)t < resid ? 1u : 0u);    // hv <= 513, exact in f16
  cum[t] = (float)hv;
  __syncthreads();

  // ---- 256-bin cumsum: rowprefix = Hm(16x16) x UpperTriOnes via WMMA ----
#if defined(__gfx1250__)
  if (t < 32) {                                     // wave 0, EXEC all ones
    const int m = t & 15, half = t >> 4;
    v16h a, bm;
    #pragma unroll
    for (int e = 0; e < 16; ++e) {
      // A 16x32 f16 layout: lane m, element e -> K = (e<8 ? e : e+8) + 8*half
      int K = (e < 8 ? e : e + 8) + half * 8;
      a[e] = (K < 16) ? (_Float16)cum[m * 16 + K] : (_Float16)0.0f;
      // B 32x16 f16 layout: lane (c=m, half), element e -> K = (e<8 ? e : e+8) + 8*half
      // B[K][c] = (K <= c, K < 16) ? 1 : 0  (inclusive upper-triangular ones)
      bm[e] = (e < 8 && half * 8 + e <= m) ? (_Float16)1.0f : (_Float16)0.0f;
    }
    v8f c0 = {};
    v8f d = __builtin_amdgcn_wmma_f32_16x16x32_f16(
        false, a, false, bm, (short)0, c0, false, false);
    #pragma unroll
    for (int v = 0; v < 8; ++v) {
      int r = v + 8 * half;                         // D row; col = m
      cum[r * 16 + m] = d[v];                       // row-wise inclusive prefix
    }
  }
#else
  if (t == 0) {
    for (int r = 0; r < 16; ++r) {
      float run = 0.0f;
      for (int c2 = 0; c2 < 16; ++c2) { run += cum[r * 16 + c2]; cum[r * 16 + c2] = run; }
    }
  }
#endif
  __syncthreads();

  // second-level scan over the 16 row totals
  if (t < 16) {
    float s = 0.0f;
    for (int q = 0; q < t; ++q) s += cum[q * 16 + 15];
    off16[t] = s;
  }
  __syncthreads();

  // LUT = clip(floor(cdf * 255/65536), 0, 255)
  float total = cum[t] + off16[t >> 4];
  float lutv  = floorf(total * (255.0f / 65536.0f));
  lutv = fminf(fmaxf(lutv, 0.0f), 255.0f);
  ws[WS_LUT + blk * NBINS + t] = lutv;
}

// ---------------- K4: apply CLAHE (bilinear LUT blend) + minmax partials -----
__global__ __launch_bounds__(256) void k_apply(const float* __restrict__ in,
                                               float* __restrict__ out,
                                               float* __restrict__ ws) {
  const float mn  = ws[WS_MN1];
  const float mx  = ws[WS_MX1];
  const float inv = 1.0f / (mx - mn);
  const float* lut = ws + WS_LUT;

  float lmn = 3.402823466e38f, lmx = -3.402823466e38f;
  const float4* in4 = (const float4*)in;
  float4* out4 = (float4*)out;

  for (unsigned i = blockIdx.x * 256u + threadIdx.x; i < NV4; i += gridDim.x * 256u) {
    float4 xv = in4[i];
    const int base = (int)(i * 4u);
    const int b    = base >> 22;                 // / (2048*2048)
    const int rem  = base & 4194303;
    const int y    = rem >> 11;                  // / 2048
    const int x0p  = rem & 2047;                 // 4-aligned, never crosses row

    const float tyf = (y + 0.5f) * (1.0f / 256.0f) - 0.5f;
    const float y0f = floorf(tyf);
    const float wy  = tyf - y0f;
    const int y0 = clampi((int)y0f, 0, 7);
    const int y1 = clampi((int)y0f + 1, 0, 7);

    float vals[4] = { xv.x, xv.y, xv.z, xv.w };
    float res[4];
    #pragma unroll
    for (int k = 0; k < 4; ++k) {
      const int x = x0p + k;
      const float txf = (x + 0.5f) * (1.0f / 256.0f) - 0.5f;
      const float x0f = floorf(txf);
      const float wx  = txf - x0f;
      const int xx0 = clampi((int)x0f, 0, 7);
      const int xx1 = clampi((int)x0f + 1, 0, 7);

      const int bin = pix_bin(vals[k], mn, inv);
      const int r0  = (b * 8 + y0) * 8;
      const int r1  = (b * 8 + y1) * 8;
      const float g00 = lut[((r0 + xx0) << 8) + bin];
      const float g01 = lut[((r0 + xx1) << 8) + bin];
      const float g10 = lut[((r1 + xx0) << 8) + bin];
      const float g11 = lut[((r1 + xx1) << 8) + bin];
      float v = (1.0f - wy) * ((1.0f - wx) * g00 + wx * g01)
              +          wy * ((1.0f - wx) * g10 + wx * g11);
      v *= (1.0f / 255.0f);
      res[k] = v;
      lmn = fminf(lmn, v);
      lmx = fmaxf(lmx, v);
    }
    out4[i] = make_float4(res[0], res[1], res[2], res[3]);
  }
  block_reduce_minmax(lmn, lmx, &ws[WS_P2MIN + blockIdx.x], &ws[WS_P2MAX + blockIdx.x]);
}

// ---------------- K5: finalize output min/max (2048 partials) ----------------
__global__ __launch_bounds__(256) void k_final2(float* __restrict__ ws) {
  const int t = threadIdx.x;
  float lmn = 3.402823466e38f, lmx = -3.402823466e38f;
  for (int i = t; i < 2048; i += 256) {
    lmn = fminf(lmn, ws[WS_P2MIN + i]);
    lmx = fmaxf(lmx, ws[WS_P2MAX + i]);
  }
  block_reduce_minmax(lmn, lmx, &ws[WS_MN2], &ws[WS_MX2]);
}

// ---------------- K6: final in-place normalize of d_out ----------------------
__global__ __launch_bounds__(256) void k_rescale(float* __restrict__ out,
                                                 const float* __restrict__ ws) {
  const float mn  = ws[WS_MN2];
  const float mx  = ws[WS_MX2];
  const float inv = 1.0f / (mx - mn);
  float4* o4 = (float4*)out;
  for (unsigned i = blockIdx.x * 256u + threadIdx.x; i < NV4; i += gridDim.x * 256u) {
    float4 v = o4[i];
    v.x = (v.x - mn) * inv;
    v.y = (v.y - mn) * inv;
    v.z = (v.z - mn) * inv;
    v.w = (v.w - mn) * inv;
    o4[i] = v;
  }
}

// ---------------- launch -----------------------------------------------------
extern "C" void kernel_launch(void* const* d_in, const int* in_sizes, int n_in,
                              void* d_out, int out_size, void* d_ws, size_t ws_size,
                              hipStream_t stream) {
  (void)in_sizes; (void)n_in; (void)out_size; (void)ws_size;
  const float* x = (const float*)d_in[0];
  float* out = (float*)d_out;
  float* ws  = (float*)d_ws;

  k_minmax_in<<<1024, 256, 0, stream>>>(x, ws);
  k_final1  <<<1,    256, 0, stream>>>(ws);
  k_tile_lut<<<NTILES, 256, 0, stream>>>(x, ws);
  k_apply   <<<2048, 256, 0, stream>>>(x, out, ws);
  k_final2  <<<1,    256, 0, stream>>>(ws);
  k_rescale <<<2048, 256, 0, stream>>>(out, ws);
}
